// TransformerDecoderLayer_21191368639302
// MI455X (gfx1250) — matrix operand order repaired and checked by reading, not compiled, not run on hardware
//
#include <hip/hip_runtime.h>

// ---------------------------------------------------------------------------
// Types for CDNA5 WMMA fragments
// ---------------------------------------------------------------------------
typedef __attribute__((ext_vector_type(16))) _Float16 v16h;
typedef __attribute__((ext_vector_type(8)))  _Float16 v8h;
typedef __attribute__((ext_vector_type(8)))  float    v8f;
typedef __attribute__((ext_vector_type(4)))  int      v4i;

union H16 { v16h v; v8h h[2]; };

#define EMBED 1024
#define HEADS 16
#define HDIM  64
#define DFF   4096
#define NB    4
#define TT    2048
#define SS    2048
#define MROWS (NB * TT)   // 8192

// ---------------------------------------------------------------------------
// CDNA5 async global->LDS copy (ASYNCcnt-tracked). Signature confirmed by the
// round-2 diagnostic: (v4i AS1*, v4i AS3*, imm offset, imm cpol).
// ---------------------------------------------------------------------------
#if defined(__HIP_DEVICE_COMPILE__) && __has_builtin(__builtin_amdgcn_global_load_async_to_lds_b128)
#define HAVE_ASYNC 1
#else
#define HAVE_ASYNC 0
#endif

__device__ __forceinline__ void copy16_async(const _Float16* g, _Float16* l) {
#if HAVE_ASYNC
  v4i* gg = (v4i*)g;   // strip const + reinterpret (generic AS)
  v4i* ll = (v4i*)l;
  __builtin_amdgcn_global_load_async_to_lds_b128(
      (__attribute__((address_space(1))) v4i*)gg,
      (__attribute__((address_space(3))) v4i*)ll, 0, 0);
#else
  *(v8h*)l = *(const v8h*)g;
#endif
}

__device__ __forceinline__ void wait_async_le4() {
#if HAVE_ASYNC
#if __has_builtin(__builtin_amdgcn_s_wait_asynccnt)
  __builtin_amdgcn_s_wait_asynccnt(4);
#else
  asm volatile("s_wait_asynccnt 0x4" ::: "memory");
#endif
#endif
}
__device__ __forceinline__ void wait_async_0() {
#if HAVE_ASYNC
#if __has_builtin(__builtin_amdgcn_s_wait_asynccnt)
  __builtin_amdgcn_s_wait_asynccnt(0);
#else
  asm volatile("s_wait_asynccnt 0x0" ::: "memory");
#endif
#endif
}

// ---------------------------------------------------------------------------
// f32 -> f16 conversion (activations enter the WMMA path)
// ---------------------------------------------------------------------------
__global__ void cvt_f32_f16(const float* __restrict__ src,
                            _Float16* __restrict__ dst, size_t n) {
  size_t i = (size_t)blockIdx.x * blockDim.x + threadIdx.x;
  if (i < n) dst[i] = (_Float16)src[i];
}

// ---------------------------------------------------------------------------
// Weight transpose+convert: W f32 [K][N] -> Wt f16 [N][K]  (32x32 LDS tile)
// Makes every GEMM B-tile a contiguous K-major copy (async-to-LDS friendly).
// ---------------------------------------------------------------------------
__global__ __launch_bounds__(256) void transpose_w(const float* __restrict__ W,
                                                   _Float16* __restrict__ Wt,
                                                   int K, int N) {
  __shared__ float tile[32][33];
  const int k0 = blockIdx.x * 32, n0 = blockIdx.y * 32;
  const int tx = threadIdx.x & 31, ty = threadIdx.x >> 5;  // 32 x 8
#pragma unroll
  for (int i = 0; i < 32; i += 8)
    tile[ty + i][tx] = W[(size_t)(k0 + ty + i) * N + n0 + tx];
  __syncthreads();
#pragma unroll
  for (int i = 0; i < 32; i += 8)
    Wt[(size_t)(n0 + ty + i) * K + k0 + tx] = (_Float16)tile[tx][ty + i];
}

// ---------------------------------------------------------------------------
// V transpose (tiled): [b, s, h*64+d] f16 -> Vt [(b*16+h)*64+d, s] f16
// ---------------------------------------------------------------------------
__global__ __launch_bounds__(256) void transpose_v(const _Float16* __restrict__ V,
                                                   _Float16* __restrict__ Vt) {
  __shared__ __align__(16) _Float16 tile[64][72];
  const int bh = blockIdx.y;
  const int b = bh >> 4, h = bh & 15;
  const int s0 = blockIdx.x * 64;
  const int r = threadIdx.x >> 2;         // 0..63
  const int c = (threadIdx.x & 3) * 16;   // 0,16,32,48
  const _Float16* src = V + ((size_t)b * SS + s0 + r) * EMBED + h * HDIM + c;
  *(v8h*)&tile[r][c]     = *(const v8h*)src;
  *(v8h*)&tile[r][c + 8] = *(const v8h*)(src + 8);
  __syncthreads();
  v8h o0, o1;
#pragma unroll
  for (int j = 0; j < 8; j++) o0[j] = tile[c + j][r];
#pragma unroll
  for (int j = 0; j < 8; j++) o1[j] = tile[c + 8 + j][r];
  _Float16* dst = Vt + ((size_t)bh * HDIM + r) * SS + s0 + c;
  *(v8h*)dst       = o0;
  *(v8h*)(dst + 8) = o1;
}

// ---------------------------------------------------------------------------
// WMMA GEMM: C[M,N] = A[M,K] * Bt[N,K]^T + bias   (A, Bt f16, K-major both)
// EPI 0: f16 out      EPI 1: relu -> f16 out      EPI 2: + f32 residual -> f32
// Block tile 128x128, 8 waves, wave tile 64x32, BK=32, double-buffered LDS
// with ASYNCcnt-tracked global->LDS copies.
// ---------------------------------------------------------------------------
template <int EPI>
__global__ __launch_bounds__(256) void gemm_f16(
    const _Float16* __restrict__ A, const _Float16* __restrict__ Bt,
    const float* __restrict__ bias, _Float16* __restrict__ C16,
    const float* __restrict__ Res, float* __restrict__ C32,
    int M, int N, int K) {
  __shared__ __align__(16) _Float16 As[2][128 * 32];
  __shared__ __align__(16) _Float16 Bs[2][128 * 32];   // [n][k] per tile

  const int tid  = threadIdx.x;
  const int lane = tid & 31;
  const int wave = tid >> 5;
  const int lh   = lane >> 4;   // lane half (0/1)
  const int l16  = lane & 15;
  const int bm = blockIdx.y * 128;
  const int bn = blockIdx.x * 128;
  const int wr = (wave >> 2) * 64;  // wave row offset inside block tile
  const int wc = (wave & 3) * 32;   // wave col offset inside block tile

  // staging addressing: thread -> (row 0..127, kchunk 0/16), 2 x 16B per tile
  const int srow = tid >> 1;
  const int skc  = (tid & 1) * 16;
  const _Float16* pa = A  + (size_t)(bm + srow) * K + skc;
  const _Float16* pb = Bt + (size_t)(bn + srow) * K + skc;
  const int soff = srow * 32 + skc;

  v8f acc[4][2];
#pragma unroll
  for (int i = 0; i < 4; i++)
#pragma unroll
    for (int j = 0; j < 2; j++) acc[i][j] = (v8f){};

  const int niter = K >> 5;

  // prologue: stage tile 0 into buffer 0
  copy16_async(pa,     &As[0][soff]);
  copy16_async(pa + 8, &As[0][soff + 8]);
  copy16_async(pb,     &Bs[0][soff]);
  copy16_async(pb + 8, &Bs[0][soff + 8]);

  for (int i = 0; i < niter; ++i) {
    const int cur = i & 1;
    if (i + 1 < niter) {
      const int nxt = cur ^ 1;
      const int kb = (i + 1) << 5;
      copy16_async(pa + kb,     &As[nxt][soff]);
      copy16_async(pa + kb + 8, &As[nxt][soff + 8]);
      copy16_async(pb + kb,     &Bs[nxt][soff]);
      copy16_async(pb + kb + 8, &Bs[nxt][soff + 8]);
      wait_async_le4();  // in-order: current tile's 4 ops complete
    } else {
      wait_async_0();
    }
    __syncthreads();

    // fragments per ISA 7.12.2 layouts
    H16 af[4], bf[2];
#pragma unroll
    for (int k = 0; k < 4; k++) {
      const int row = wr + k * 16 + l16;
      af[k].h[0] = *(const v8h*)(&As[cur][row * 32 + lh * 8]);       // K 0..7 / 8..15
      af[k].h[1] = *(const v8h*)(&As[cur][row * 32 + 16 + lh * 8]);  // K 16..23 / 24..31
    }
#pragma unroll
    for (int j = 0; j < 2; j++) {
      const int col = wc + j * 16 + l16;
      bf[j].h[0] = *(const v8h*)(&Bs[cur][col * 32 + lh * 8]);       // K 0..7 / 8..15
      bf[j].h[1] = *(const v8h*)(&Bs[cur][col * 32 + 16 + lh * 8]);  // K 16..23 / 24..31
    }
#pragma unroll
    for (int k = 0; k < 4; k++)
#pragma unroll
      for (int j = 0; j < 2; j++)
        acc[k][j] = __builtin_amdgcn_wmma_f32_16x16x32_f16(
            false, af[k].v, false, bf[j].v, (short)0, acc[k][j], false, false);
    __syncthreads();  // protect buffer `cur` before it is restaged at i+2
  }

  // epilogue: C layout = lane holds column (l16), rows lh*8+e
#pragma unroll
  for (int i = 0; i < 4; i++) {
#pragma unroll
    for (int j = 0; j < 2; j++) {
      const int col = bn + wc + j * 16 + l16;
      const float bv = bias[col];
#pragma unroll
      for (int e = 0; e < 8; e++) {
        const int row = bm + wr + i * 16 + lh * 8 + e;
        const size_t idx = (size_t)row * N + col;
        float val = acc[i][j][e] + bv;
        if (EPI == 1) val = val > 0.f ? val : 0.f;
        if (EPI == 2) {
          C32[idx] = val + Res[idx];
        } else {
          C16[idx] = (_Float16)val;
        }
      }
    }
  }
}

// ---------------------------------------------------------------------------
// Flash attention, one wave per 16-query tile, f32 online softmax.
// Q,K: [b*T + t, h*64 + d] f16.  Vt: [(b*16+h)*64 + d, s] f16.
// ---------------------------------------------------------------------------
template <bool CAUSAL>
__global__ __launch_bounds__(256) void flash_attn(
    const _Float16* __restrict__ Q, const _Float16* __restrict__ Kk,
    const _Float16* __restrict__ Vt, _Float16* __restrict__ Y) {
  __shared__ __align__(16) _Float16 Pst[8][16 * 32];  // per-wave P staging

  const int lane = threadIdx.x & 31;
  const int wave = threadIdx.x >> 5;
  const int lh   = lane >> 4;
  const int l16  = lane & 15;
  const int bh = blockIdx.y;           // b*16 + h
  const int b = bh >> 4, h = bh & 15;
  const int qbase = (blockIdx.x * 8 + wave) * 16;

  // Q fragments (A-layout, 16x32 per head-dim chunk)
  H16 qf[2];
  {
    const _Float16* qp = Q + ((size_t)b * TT + qbase + l16) * EMBED + h * HDIM;
#pragma unroll
    for (int c = 0; c < 2; c++) {
      qf[c].h[0] = *(const v8h*)(qp + c * 32 + lh * 8);
      qf[c].h[1] = *(const v8h*)(qp + c * 32 + 16 + lh * 8);
    }
  }

  v8f o[4];
#pragma unroll
  for (int t = 0; t < 4; t++) o[t] = (v8f){};
  float rmax[8], rsum[8];
#pragma unroll
  for (int e = 0; e < 8; e++) { rmax[e] = -3.0e38f; rsum[e] = 0.f; }

  const float LOG2E = 1.4426950408889634f;
  const int kend = CAUSAL ? (qbase + 16) : SS;

  for (int kb = 0; kb < kend; kb += 32) {
    // prefetch next K/V tiles (global_prefetch_b8; speculative, safe)
    if (kb + 64 <= kend) {
      __builtin_prefetch(Kk + ((size_t)b * SS + kb + 32 + lane) * EMBED + h * HDIM, 0, 1);
      __builtin_prefetch(Vt + ((size_t)bh * HDIM + lane * 2) * SS + kb + 32, 0, 1);
    }

    // ---- S = Q K^T : B-frag K-dim = head_dim (contiguous in K buffer)
    v8f sc[2];
#pragma unroll
    for (int nk = 0; nk < 2; nk++) {
      v8f a = (v8f){};
#pragma unroll
      for (int c = 0; c < 2; c++) {
        const _Float16* kp = Kk + ((size_t)b * SS + kb + nk * 16 + l16) * EMBED +
                             h * HDIM + c * 32 + lh * 8;
        H16 bf;
        bf.h[0] = *(const v8h*)(kp);
        bf.h[1] = *(const v8h*)(kp + 16);
        a = __builtin_amdgcn_wmma_f32_16x16x32_f16(
            false, qf[c].v, false, bf.v, (short)0, a, false, false);
      }
      sc[nk] = a;
    }

    // ---- scale + mask + online softmax (row = qbase + lh*8 + e, col = key)
    float mnew[8];
#pragma unroll
    for (int e = 0; e < 8; e++) {
      const int qrow = qbase + lh * 8 + e;
#pragma unroll
      for (int nk = 0; nk < 2; nk++) {
        const int key = kb + nk * 16 + l16;
        float sv = sc[nk][e] * 0.125f;  // 1/sqrt(64)
        if (CAUSAL && key > qrow) sv = -3.0e38f;
        sc[nk][e] = sv;
      }
      float mx = fmaxf(sc[0][e], sc[1][e]);
#pragma unroll
      for (int off = 1; off < 16; off <<= 1) mx = fmaxf(mx, __shfl_xor(mx, off, 32));
      mnew[e] = fmaxf(rmax[e], mx);
    }
#pragma unroll
    for (int e = 0; e < 8; e++) {
      const float corr = exp2f((rmax[e] - mnew[e]) * LOG2E);
      rmax[e] = mnew[e];
      float ps = 0.f;
#pragma unroll
      for (int nk = 0; nk < 2; nk++) {
        const float p = exp2f((sc[nk][e] - mnew[e]) * LOG2E);
        sc[nk][e] = p;
        ps += p;
      }
#pragma unroll
      for (int off = 1; off < 16; off <<= 1) ps += __shfl_xor(ps, off, 32);
      rsum[e] = rsum[e] * corr + ps;
#pragma unroll
      for (int t = 0; t < 4; t++) o[t][e] *= corr;
    }

    // ---- C-layout -> A-layout via 1KB per-wave LDS round-trip
    _Float16* pst = Pst[wave];
#pragma unroll
    for (int nk = 0; nk < 2; nk++)
#pragma unroll
      for (int e = 0; e < 8; e++)
        pst[(lh * 8 + e) * 32 + nk * 16 + l16] = (_Float16)sc[nk][e];
    H16 pf;
    pf.h[0] = *(const v8h*)(&pst[l16 * 32 + lh * 8]);
    pf.h[1] = *(const v8h*)(&pst[l16 * 32 + 16 + lh * 8]);

    // ---- O += P V : B-frag K-dim = keys, contiguous in Vt
#pragma unroll
    for (int t = 0; t < 4; t++) {
      const _Float16* vp = Vt + ((size_t)bh * HDIM + t * 16 + l16) * SS + kb + lh * 8;
      H16 vf;
      vf.h[0] = *(const v8h*)(vp);
      vf.h[1] = *(const v8h*)(vp + 16);
      o[t] = __builtin_amdgcn_wmma_f32_16x16x32_f16(
          false, pf.v, false, vf.v, (short)0, o[t], false, false);
    }
  }

  // ---- normalize + store y as f16 (feeds output-projection GEMM)
#pragma unroll
  for (int e = 0; e < 8; e++) {
    const float inv = 1.f / rsum[e];
#pragma unroll
    for (int t = 0; t < 4; t++) o[t][e] *= inv;
  }
#pragma unroll
  for (int t = 0; t < 4; t++)
#pragma unroll
    for (int e = 0; e < 8; e++)
      Y[((size_t)b * TT + qbase + lh * 8 + e) * EMBED + h * HDIM + t * 16 + l16] =
          (_Float16)o[t][e];
}

// ---------------------------------------------------------------------------
// LayerNorm over 1024 cols; one block per row; writes f32 (+ optional f16)
// ---------------------------------------------------------------------------
template <bool WRITE_H>
__global__ __launch_bounds__(256) void layernorm_k(
    const float* __restrict__ X, const float* __restrict__ g,
    const float* __restrict__ bta, float* __restrict__ Yf,
    _Float16* __restrict__ Yh) {
  __shared__ float red[16];
  const int row = blockIdx.x;
  const float* x = X + (size_t)row * EMBED;
  float v4[4];
  float s = 0.f, s2 = 0.f;
#pragma unroll
  for (int i = 0; i < 4; i++) {
    const float val = x[threadIdx.x + i * 256];
    v4[i] = val; s += val; s2 += val * val;
  }
#pragma unroll
  for (int off = 1; off < 32; off <<= 1) {
    s  += __shfl_xor(s,  off, 32);
    s2 += __shfl_xor(s2, off, 32);
  }
  if ((threadIdx.x & 31) == 0) {
    red[threadIdx.x >> 5]       = s;
    red[8 + (threadIdx.x >> 5)] = s2;
  }
  __syncthreads();
  s = 0.f; s2 = 0.f;
#pragma unroll
  for (int w = 0; w < 8; w++) { s += red[w]; s2 += red[8 + w]; }
  const float mu  = s * (1.f / EMBED);
  const float var = s2 * (1.f / EMBED) - mu * mu;
  const float inv = __frsqrt_rn(var + 1e-5f);
#pragma unroll
  for (int i = 0; i < 4; i++) {
    const int c = threadIdx.x + i * 256;
    const float y = (v4[i] - mu) * inv * g[c] + bta[c];
    Yf[(size_t)row * EMBED + c] = y;
    if (WRITE_H) Yh[(size_t)row * EMBED + c] = (_Float16)y;
  }
}

// ---------------------------------------------------------------------------
// Host-side orchestration
// ---------------------------------------------------------------------------
extern "C" void kernel_launch(void* const* d_in, const int* in_sizes, int n_in,
                              void* d_out, int out_size, void* d_ws, size_t ws_size,
                              hipStream_t stream) {
  const float* tgt    = (const float*)d_in[0];
  const float* memory = (const float*)d_in[1];
  // d_in[2] = tgt_mask (causal, handled analytically)
  const float* W[8];  const float* Wb[8];
  for (int i = 0; i < 8; i++) { W[i] = (const float*)d_in[3 + 2 * i]; Wb[i] = (const float*)d_in[4 + 2 * i]; }
  const float* lin1_W = (const float*)d_in[19]; const float* lin1_b = (const float*)d_in[20];
  const float* lin2_W = (const float*)d_in[21]; const float* lin2_b = (const float*)d_in[22];
  const float* ln_g[3] = {(const float*)d_in[23], (const float*)d_in[25], (const float*)d_in[27]};
  const float* ln_b[3] = {(const float*)d_in[24], (const float*)d_in[26], (const float*)d_in[28]};

  const size_t MD = (size_t)MROWS * EMBED;        // 8M elements
  const size_t DD = (size_t)EMBED * EMBED;        // 1M
  const size_t DF = (size_t)EMBED * DFF;          // 4M
  const size_t MF = (size_t)MROWS * DFF;          // 32M

  char* ws = (char*)d_ws;
  size_t off = 0;
  auto alloc = [&](size_t bytes) {
    void* p = ws + off;
    off += (bytes + 255) & ~(size_t)255;
    return p;
  };
  _Float16* hx   = (_Float16*)alloc(MD * 2);
  _Float16* hmem = (_Float16*)alloc(MD * 2);
  _Float16* Wt[8];
  for (int i = 0; i < 8; i++) Wt[i] = (_Float16*)alloc(DD * 2);   // [N=1024][K=1024]
  _Float16* w1t = (_Float16*)alloc(DF * 2);                       // [N=4096][K=1024]
  _Float16* w2t = (_Float16*)alloc(DF * 2);                       // [N=1024][K=4096]
  _Float16* qb  = (_Float16*)alloc(MD * 2);
  _Float16* kb  = (_Float16*)alloc(MD * 2);
  _Float16* vb  = (_Float16*)alloc(MD * 2);
  _Float16* vt  = (_Float16*)alloc(MD * 2);
  _Float16* yb  = (_Float16*)alloc(MD * 2);
  float*    sbuf = (float*)alloc(MD * 4);
  float*    x1f  = (float*)alloc(MD * 4);
  _Float16* x1h  = (_Float16*)alloc(MD * 2);
  float*    x2f  = (float*)alloc(MD * 4);
  _Float16* x2h  = (_Float16*)alloc(MD * 2);
  _Float16* ffh  = (_Float16*)alloc(MF * 2);
  (void)ws_size; (void)in_sizes; (void)n_in; (void)out_size;

  const dim3 blk(256);

  // activations f32 -> f16
  cvt_f32_f16<<<(unsigned)((MD + 255) / 256), blk, 0, stream>>>(tgt, hx, MD);
  cvt_f32_f16<<<(unsigned)((MD + 255) / 256), blk, 0, stream>>>(memory, hmem, MD);
  // weights: transpose + convert -> K-major f16
  for (int i = 0; i < 8; i++)
    transpose_w<<<dim3(EMBED / 32, EMBED / 32), blk, 0, stream>>>(W[i], Wt[i], EMBED, EMBED);
  transpose_w<<<dim3(EMBED / 32, DFF / 32), blk, 0, stream>>>(lin1_W, w1t, EMBED, DFF);
  transpose_w<<<dim3(DFF / 32, EMBED / 32), blk, 0, stream>>>(lin2_W, w2t, DFF, EMBED);

  const dim3 gD(EMBED / 128, MROWS / 128);   // (8, 64)
  const dim3 gF(DFF / 128, MROWS / 128);     // (32, 64)
  const dim3 gA(TT / 128, NB * HEADS);       // (16, 64)
  const dim3 gV(SS / 64, NB * HEADS);        // (32, 64)

  // ---- self-attention ----
  gemm_f16<0><<<gD, blk, 0, stream>>>(hx, Wt[0], Wb[0], qb, nullptr, nullptr, MROWS, EMBED, EMBED);
  gemm_f16<0><<<gD, blk, 0, stream>>>(hx, Wt[1], Wb[1], kb, nullptr, nullptr, MROWS, EMBED, EMBED);
  gemm_f16<0><<<gD, blk, 0, stream>>>(hx, Wt[2], Wb[2], vb, nullptr, nullptr, MROWS, EMBED, EMBED);
  transpose_v<<<gV, blk, 0, stream>>>(vb, vt);
  flash_attn<true><<<gA, blk, 0, stream>>>(qb, kb, vt, yb);
  gemm_f16<2><<<gD, blk, 0, stream>>>(yb, Wt[3], Wb[3], nullptr, tgt, sbuf, MROWS, EMBED, EMBED);
  layernorm_k<true><<<MROWS, blk, 0, stream>>>(sbuf, ln_g[0], ln_b[0], x1f, x1h);

  // ---- cross-attention ----
  gemm_f16<0><<<gD, blk, 0, stream>>>(x1h,  Wt[4], Wb[4], qb, nullptr, nullptr, MROWS, EMBED, EMBED);
  gemm_f16<0><<<gD, blk, 0, stream>>>(hmem, Wt[5], Wb[5], kb, nullptr, nullptr, MROWS, EMBED, EMBED);
  gemm_f16<0><<<gD, blk, 0, stream>>>(hmem, Wt[6], Wb[6], vb, nullptr, nullptr, MROWS, EMBED, EMBED);
  transpose_v<<<gV, blk, 0, stream>>>(vb, vt);
  flash_attn<false><<<gA, blk, 0, stream>>>(qb, kb, vt, yb);
  gemm_f16<2><<<gD, blk, 0, stream>>>(yb, Wt[7], Wb[7], nullptr, x1f, sbuf, MROWS, EMBED, EMBED);
  layernorm_k<true><<<MROWS, blk, 0, stream>>>(sbuf, ln_g[1], ln_b[1], x2f, x2h);

  // ---- FFN ----
  gemm_f16<1><<<gF, blk, 0, stream>>>(x2h, w1t, lin1_b, ffh, nullptr, nullptr, MROWS, DFF, EMBED);
  gemm_f16<2><<<gD, blk, 0, stream>>>(ffh, w2t, lin2_b, nullptr, x2f, sbuf, MROWS, EMBED, DFF);
  layernorm_k<false><<<MROWS, blk, 0, stream>>>(sbuf, ln_g[2], ln_b[2], (float*)d_out, nullptr);
}
